// SeqAttentionBlock_39591008534668
// MI455X (gfx1250) — matrix operand order repaired and checked by reading, not compiled, not run on hardware
//
#include <hip/hip_runtime.h>
#include <hip/hip_bf16.h>

typedef _Float16 v16h __attribute__((ext_vector_type(16)));
typedef _Float16 h8   __attribute__((ext_vector_type(8)));
typedef _Float16 h2   __attribute__((ext_vector_type(2)));
typedef float    v8f  __attribute__((ext_vector_type(8)));

#define BB 8
#define TT 512
#define MM 16
#define DD 128
#define PP 256
#define HH 8
#define EE 32

// LDS layout (units: _Float16 elements). Strides chosen so every 16-byte
// vector access is aligned and the 16 lanes of a half-wave hit distinct banks.
#define WT_STRIDE 136   // 272B = 17*16B; ≡ 4 banks mod 64
#define K_STRIDE  40    // 80B  = 5*16B;  20 banks/row, n*20 mod 64 distinct
#define VT_STRIDE 520   // 1040B = 65*16B; ≡ 4 banks mod 64
#define QS_STRIDE 40

#define OFF_WT 0
#define N_WT   (3 * 32 * WT_STRIDE)
#define OFF_K  (N_WT)
#define N_K    (TT * K_STRIDE)
#define OFF_VT (OFF_K + N_K)
#define N_VT   (EE * VT_STRIDE)
#define OFF_QS (OFF_VT + N_VT)
#define N_QS   (8 * 16 * QS_STRIDE)
#define OFF_PS (OFF_QS + N_QS)
#define SMEM_HALFS (OFF_PS + N_QS)
#define SMEM_BYTES (SMEM_HALFS * 2 + TT)   // + 512B key mask

#define WMMA16(A, B, C) \
  __builtin_amdgcn_wmma_f32_16x16x32_f16(false, (A), false, (B), (short)0, (C), false, false)

__device__ __forceinline__ v16h mk16(h8 lo, h8 hi) {
  v16h r;
#pragma unroll
  for (int j = 0; j < 8; ++j) { r[j] = lo[j]; r[8 + j] = hi[j]; }
  return r;
}

// B-fragment: lane reads 16 contiguous f16 (column l16, K = half*16 .. +15)
__device__ __forceinline__ v16h ldsB16(const _Float16* p) {
  return mk16(*(const h8*)p, *(const h8*)(p + 8));
}

// A-fragment from LDS scratch: row = l16; elems 0..15 = K 2*off..2*off+15
// (off = half*8)
__device__ __forceinline__ v16h ldsA16(const _Float16* row, int off) {
  return mk16(*(const h8*)(row + 2 * off), *(const h8*)(row + 2 * off + 8));
}

// A-fragment straight from global fp32 X row, converted to f16 in-register.
__device__ __forceinline__ v16h xA16(const float* xrow, int kbase, int off) {
  const float4* p0 = (const float4*)(xrow + kbase + 2 * off);
  const float4* p1 = (const float4*)(xrow + kbase + 2 * off + 8);
  float4 a0 = p0[0], a1 = p0[1], b0 = p1[0], b1 = p1[1];
  v16h r;
  r[0] = (_Float16)a0.x;  r[1] = (_Float16)a0.y;  r[2] = (_Float16)a0.z;  r[3] = (_Float16)a0.w;
  r[4] = (_Float16)a1.x;  r[5] = (_Float16)a1.y;  r[6] = (_Float16)a1.z;  r[7] = (_Float16)a1.w;
  r[8] = (_Float16)b0.x;  r[9] = (_Float16)b0.y;  r[10] = (_Float16)b0.z; r[11] = (_Float16)b0.w;
  r[12] = (_Float16)b1.x; r[13] = (_Float16)b1.y; r[14] = (_Float16)b1.z; r[15] = (_Float16)b1.w;
  return r;
}

__global__ __launch_bounds__(256)
void seq_attn_fused(const float* __restrict__ inp, const unsigned char* __restrict__ maskp,
                    const float* __restrict__ Wq, const float* __restrict__ bq,
                    const float* __restrict__ Wk, const float* __restrict__ bk,
                    const float* __restrict__ Wv, const float* __restrict__ bv,
                    float* __restrict__ out) {
  extern __shared__ _Float16 sh[];
  _Float16* sWt = sh + OFF_WT;   // [3][32][WT_STRIDE]  W^T slices (e, d), f16
  _Float16* sK  = sh + OFF_K;    // [512][K_STRIDE]     K (t, e), f16
  _Float16* sVt = sh + OFF_VT;   // [32][VT_STRIDE]     V^T (e, pos(t)), f16 (permuted in 32-chunks)
  _Float16* sQ  = sh + OFF_QS;   // [8][16][QS_STRIDE]  per-wave Q tile scratch
  _Float16* sP  = sh + OFF_PS;   // [8][16][QS_STRIDE]  per-wave P tile scratch (permuted K order)
  unsigned char* smask = (unsigned char*)(sh + SMEM_HALFS);

  const int tid  = threadIdx.x;
  const int lane = tid & 31;
  const int wave = tid >> 5;
  const int l16  = lane & 15;
  const int half = lane >> 4;

  const int wg = blockIdx.x;
  const int h  = wg & 7;
  const int m  = (wg >> 3) & 15;
  const int b  = wg >> 7;

  const float* Xbase = inp + ((size_t)b * TT * MM + (size_t)m) * DD;  // row t: + t*MM*DD

  // ---- Stage W head-slices (transposed, f16) and the key mask ----
  for (int idx = tid; idx < DD * EE; idx += 256) {
    const int d = idx >> 5;
    const int e = idx & 31;
    const size_t src = (size_t)d * PP + h * EE + e;
    const int dst = e * WT_STRIDE + d;
    sWt[dst]                      = (_Float16)Wq[src];
    sWt[32 * WT_STRIDE + dst]     = (_Float16)Wk[src];
    sWt[2 * 32 * WT_STRIDE + dst] = (_Float16)Wv[src];
  }
  for (int idx = tid; idx < TT; idx += 256)
    smask[idx] = maskp[((size_t)b * TT + idx) * MM + m];
  __syncthreads();

  // ---- K and V projections into LDS. One t-tile job computes all four
  // 16x16 output tiles (K/V x e0/e1) from a single X A-fragment per kd. ----
  const _Float16* wK = sWt + 32 * WT_STRIDE;
  const _Float16* wV = sWt + 2 * 32 * WT_STRIDE;
  for (int job = wave; job < 32; job += 8) {
    const int t0 = job << 4;
    const float* xrow = Xbase + (size_t)(t0 + l16) * (MM * DD);
    v8f aK0 = {}, aK1 = {}, aV0 = {}, aV1 = {};
#pragma unroll
    for (int kd = 0; kd < 4; ++kd) {
      v16h a = xA16(xrow, kd * 32, half * 8);
      const int wo = kd * 32 + half * 16;
      aK0 = WMMA16(a, ldsB16(wK + l16 * WT_STRIDE + wo), aK0);
      aK1 = WMMA16(a, ldsB16(wK + (16 + l16) * WT_STRIDE + wo), aK1);
      aV0 = WMMA16(a, ldsB16(wV + l16 * WT_STRIDE + wo), aV0);
      aV1 = WMMA16(a, ldsB16(wV + (16 + l16) * WT_STRIDE + wo), aV1);
    }
    const float bk0 = bk[h * EE + l16],      bk1 = bk[h * EE + 16 + l16];
    const float bv0 = bv[h * EE + l16],      bv1 = bv[h * EE + 16 + l16];
    const int sub   = (t0 >> 4) & 1;         // which half of the 32-chunk
    const int chunk = t0 & ~31;
#pragma unroll
    for (int i = 0; i < 8; ++i) {
      const int r = i + half * 8;
      sK[(t0 + r) * K_STRIDE + l16]      = (_Float16)(aK0[i] + bk0);
      sK[(t0 + r) * K_STRIDE + 16 + l16] = (_Float16)(aK1[i] + bk1);
      const int vt = chunk + 2 * r + sub;    // permuted position within chunk
      sVt[l16 * VT_STRIDE + vt]        = (_Float16)(aV0[i] + bv0);
      sVt[(16 + l16) * VT_STRIDE + vt] = (_Float16)(aV1[i] + bv1);
    }
  }
  __syncthreads();

  // Per-lane key-mask register: bit k = mask for column k*16 + l16.
  unsigned mbits = 0;
#pragma unroll
  for (int k = 0; k < 32; ++k)
    mbits |= (unsigned)(smask[k * 16 + l16] != 0) << k;

  const float softscale = 0.17677669529663687f;  // 1/sqrt(E)
  const v8f vzero = {};

  // ---- Each wave owns q-tiles: qt = wave, wave+8, ... (uniform, EXEC full) ----
  for (int qt = wave; qt < 32; qt += 8) {
    const int t0 = qt << 4;
    const float* xrow = Xbase + (size_t)(t0 + l16) * (MM * DD);

    // Q tile (biased, pre-scaled) -> per-wave scratch (one A-frag load per kd).
    {
      v8f q0 = {}, q1 = {};
#pragma unroll
      for (int kd = 0; kd < 4; ++kd) {
        v16h a = xA16(xrow, kd * 32, half * 8);
        const int wo = kd * 32 + half * 16;
        q0 = WMMA16(a, ldsB16(sWt + l16 * WT_STRIDE + wo), q0);
        q1 = WMMA16(a, ldsB16(sWt + (16 + l16) * WT_STRIDE + wo), q1);
      }
      const float bq0 = bq[h * EE + l16], bq1 = bq[h * EE + 16 + l16];
#pragma unroll
      for (int i = 0; i < 8; ++i) {
        const int r = (wave * 16 + i + half * 8) * QS_STRIDE;
        sQ[r + l16]      = (_Float16)((q0[i] + bq0) * softscale);
        sQ[r + 16 + l16] = (_Float16)((q1[i] + bq1) * softscale);
      }
    }
    v16h Qa = ldsA16(sQ + (wave * 16 + l16) * QS_STRIDE, half * 8);

    // Pass 1: masked row max. S tile = one WMMA (K = E = 32).
    float mx[8];
#pragma unroll
    for (int i = 0; i < 8; ++i) mx[i] = -1e30f;
    for (int st = 0; st < TT; st += 16) {
      v16h Kb = ldsB16(sK + (st + l16) * K_STRIDE + half * 16);
      v8f s = WMMA16(Qa, Kb, vzero);
      const bool keep = (mbits >> (st >> 4)) & 1u;
#pragma unroll
      for (int i = 0; i < 8; ++i) mx[i] = fmaxf(mx[i], keep ? s[i] : -1e30f);
    }
#pragma unroll
    for (int d = 1; d < 16; d <<= 1) {
#pragma unroll
      for (int i = 0; i < 8; ++i) mx[i] = fmaxf(mx[i], __shfl_xor(mx[i], d, 16));
    }

    // Pass 2: recompute S, exponentiate, stage P packed (permuted K order),
    // accumulate O = P.V over 32-wide key blocks.
    float sum[8] = {0.f, 0.f, 0.f, 0.f, 0.f, 0.f, 0.f, 0.f};
    v8f o0 = {}, o1 = {};
    for (int sb = 0; sb < TT; sb += 32) {
      v16h Kb0 = ldsB16(sK + (sb + l16) * K_STRIDE + half * 16);
      v8f s0 = WMMA16(Qa, Kb0, vzero);
      v16h Kb1 = ldsB16(sK + (sb + 16 + l16) * K_STRIDE + half * 16);
      v8f s1 = WMMA16(Qa, Kb1, vzero);
      const bool k0 = (mbits >> (sb >> 4)) & 1u;
      const bool k1 = (mbits >> ((sb >> 4) + 1)) & 1u;
#pragma unroll
      for (int i = 0; i < 8; ++i) {
        const float p0 = k0 ? __expf(s0[i] - mx[i]) : 0.f;
        const float p1 = k1 ? __expf(s1[i] - mx[i]) : 0.f;
        sum[i] += p0 + p1;
        h2 pr; pr[0] = (_Float16)p0; pr[1] = (_Float16)p1;
        *(h2*)&sP[(wave * 16 + i + half * 8) * QS_STRIDE + 2 * l16] = pr;
      }
      v16h Pa = ldsA16(sP + (wave * 16 + l16) * QS_STRIDE, half * 8);
      v16h V0 = ldsB16(sVt + l16 * VT_STRIDE + sb + half * 16);
      o0 = WMMA16(Pa, V0, o0);
      v16h V1 = ldsB16(sVt + (16 + l16) * VT_STRIDE + sb + half * 16);
      o1 = WMMA16(Pa, V1, o1);
    }
#pragma unroll
    for (int d = 1; d < 16; d <<= 1) {
#pragma unroll
      for (int i = 0; i < 8; ++i) sum[i] += __shfl_xor(sum[i], d, 16);
    }

    // Normalize (all-keys-masked rows get sum==0 -> output 0) and store.
    float* obase = out + (((size_t)b * TT + t0) * MM + m) * PP + h * EE;
#pragma unroll
    for (int i = 0; i < 8; ++i) {
      const float rs = sum[i] > 0.f ? 1.f / sum[i] : 0.f;
      const int row = i + half * 8;
      obase[(size_t)row * (MM * PP) + l16]      = o0[i] * rs;
      obase[(size_t)row * (MM * PP) + 16 + l16] = o1[i] * rs;
    }
  }
}

extern "C" void kernel_launch(void* const* d_in, const int* in_sizes, int n_in,
                              void* d_out, int out_size, void* d_ws, size_t ws_size,
                              hipStream_t stream) {
  const float*         inp  = (const float*)d_in[0];
  const unsigned char* mask = (const unsigned char*)d_in[1];
  const float*         Wq   = (const float*)d_in[2];
  const float*         bq   = (const float*)d_in[3];
  const float*         Wk   = (const float*)d_in[4];
  const float*         bk   = (const float*)d_in[5];
  const float*         Wv   = (const float*)d_in[6];
  const float*         bv   = (const float*)d_in[7];
  float*               out  = (float*)d_out;
  (void)in_sizes; (void)n_in; (void)out_size; (void)d_ws; (void)ws_size;

  seq_attn_fused<<<dim3(BB * MM * HH), dim3(256), SMEM_BYTES, stream>>>(
      inp, mask, Wq, bq, Wk, bk, Wv, bv, out);
}
